// KalmanLSTM_38001870635654
// MI455X (gfx1250) — matrix operand order repaired and checked by reading, not compiled, not run on hardware
//
#include <hip/hip_runtime.h>
#include <hip/hip_bf16.h>
#include <math.h>

typedef __attribute__((ext_vector_type(16))) _Float16 v16h;
typedef __attribute__((ext_vector_type(8)))  float    v8f;

#define FEATN   128
#define NGATE   512           // 4*FEAT
#define KTOT    256           // feat(128) + hx(128)
#define MB      32            // batch rows per block
#define NTHREADS 256
#define THIST   16
#define BATCHN  8192
#define DTC     0.2f

// ---------------------------------------------------------------------------
// Prep kernel: pack [W_ih ; W_hh] -> f16 Wcat[512][256], biasC = b_ih + b_hh
// ---------------------------------------------------------------------------
__global__ void kprep_pack(const float* __restrict__ W_ih,
                           const float* __restrict__ W_hh,
                           const float* __restrict__ b_ih,
                           const float* __restrict__ b_hh,
                           _Float16* __restrict__ Wcat,
                           float* __restrict__ biasC) {
    int idx = blockIdx.x * blockDim.x + threadIdx.x;
    if (idx < NGATE * KTOT) {
        int n = idx / KTOT, k = idx % KTOT;
        float v = (k < FEATN) ? W_ih[n * FEATN + k] : W_hh[n * FEATN + (k - FEATN)];
        Wcat[idx] = (_Float16)v;
    }
    if (idx < NGATE) biasC[idx] = b_ih[idx] + b_hh[idx];
}

__device__ __forceinline__ float sigm(float x) { return 1.0f / (1.0f + expf(-x)); }

// A fragment (16x32 f16): lanes 0-15 -> rows 0-15 K-chunks {k0..k0+7, k0+16..k0+23},
// lanes 16-31 -> same rows, K shifted by +8 / +24.
__device__ __forceinline__ v16h load_a(const _Float16* __restrict__ base,
                                       int mt, int k0, int lane) {
    int row = mt * 16 + (lane & 15);
    int kc  = k0 + ((lane >> 4) << 3);
    const _Float16* p = base + row * KTOT + kc;
    v16h a;
#pragma unroll
    for (int i = 0; i < 8; ++i) { a[i] = p[i]; a[i + 8] = p[i + 16]; }
    return a;
}

// B fragment (32x16 f16): lane n = lane&15 column, K-chunks {k0..k0+7, k0+16..k0+23},
// lanes 16-31 K shifted by +8 / +24 (matches load_a); Wcat is [N][K] row-major.
__device__ __forceinline__ v16h load_b(const _Float16* __restrict__ Wcat,
                                       int n0, int k0, int lane) {
    int col = n0 + (lane & 15);
    int kr  = k0 + ((lane >> 4) << 3);
    const _Float16* p = Wcat + col * KTOT + kr;
    v16h b;
#pragma unroll
    for (int i = 0; i < 8; ++i) { b[i] = p[i]; b[i + 8] = p[i + 16]; }
    return b;
}

// ---------------------------------------------------------------------------
// Main persistent kernel: 1 block handles MB batch rows for all time steps.
// ---------------------------------------------------------------------------
__launch_bounds__(NTHREADS, 1)
__global__ void kalman_lstm_main(
    const float* __restrict__ hist,       // [16][8192][2]
    const float* __restrict__ psx, const float* __restrict__ psy,
    const float* __restrict__ vsx, const float* __restrict__ vsy,
    const float* __restrict__ asx, const float* __restrict__ asy,
    const float* __restrict__ jerk,       // [1][2]
    const float* __restrict__ coefG,      // [6][6]
    const float* __restrict__ GR,         // [2][2]
    const float* __restrict__ W_cf,       // [128][6]
    const float* __restrict__ b_cf,       // [128]
    const _Float16* __restrict__ Wcat,    // [512][256] packed
    const float* __restrict__ biasC,      // [512]
    const float* __restrict__ W_out,      // [6][128]
    const float* __restrict__ b_out,      // [6]
    const int* __restrict__ len_pred_p,
    float* __restrict__ out)              // [LP][8192][5]
{
    __shared__ _Float16 sA[MB][KTOT];     // [feat | hx_f16]   16 KB
    __shared__ float sHx[MB][FEATN];      // 16 KB
    __shared__ float sCx[MB][FEATN];      // 16 KB
    __shared__ float sX[MB][6];
    __shared__ float sCmd[MB][6];
    __shared__ float sRho[36];
    __shared__ float sR[4];
    __shared__ float sGJ[12];

    const int tid  = threadIdx.x;
    const int lane = tid & 31;
    const int wv   = tid >> 5;
    const int b0   = blockIdx.x * MB;
    const int LP   = len_pred_p[0];
    const int TOTAL = (THIST - 1) + LP;

    const float dt = DTC, h2 = DTC * DTC * 0.5f;
    const float g0c = DTC * DTC * DTC / 6.0f, g1c = h2, g2c = DTC;

    // ---- uniform constants into LDS ----
    if (tid < 36) {
        int i = tid / 6, j = tid % 6;
        float s = 0.f;
#pragma unroll
        for (int k = 0; k < 6; ++k) s += coefG[i * 6 + k] * coefG[j * 6 + k];
        sRho[tid] = s;
    }
    if (tid < 4) {
        int i = tid / 2, j = tid % 2;
        sR[tid] = GR[i * 2 + 0] * GR[j * 2 + 0] + GR[i * 2 + 1] * GR[j * 2 + 1];
    }
    if (tid < 12) {
        int i = tid / 2, j = tid % 2;
        float col = (i % 3 == 0) ? g0c : ((i % 3 == 1) ? g1c : g2c);
        float Gv = ((j == 0 && i < 3) || (j == 1 && i >= 3)) ? col : 0.f;
        sGJ[tid] = Gv * jerk[j];
    }

    // ---- zero recurrent state ----
    for (int idx = tid; idx < MB * FEATN; idx += NTHREADS) {
        int m = idx / FEATN, j = idx % FEATN;
        sHx[m][j] = 0.f;
        sCx[m][j] = 0.f;
        sA[m][FEATN + j] = (_Float16)0.f;
    }

    // ---- Kalman per-row state in registers (wave 0, one lane per row) ----
    float X[6], P[36];
    if (tid < MB) {
        int b = b0 + tid;
        float z0x = hist[(0 * BATCHN + b) * 2 + 0];
        float z0y = hist[(0 * BATCHN + b) * 2 + 1];
        float z1x = hist[(1 * BATCHN + b) * 2 + 0];
        float z1y = hist[(1 * BATCHN + b) * 2 + 1];
        // replicate reference init exactly (state3 gets v_y)
        X[0] = z0x; X[1] = (z1x - z0x) / dt; X[2] = 0.f;
        X[3] = (z1y - z0y) / dt; X[4] = 0.f; X[5] = 0.f;
#pragma unroll
        for (int i = 0; i < 36; ++i) P[i] = 0.f;
        P[0]  = psx[0] * psx[0];
        P[7]  = vsx[0] * vsx[0];
        P[14] = asx[0] * asx[0];
        P[21] = psy[0] * psy[0];
        P[28] = vsy[0] * vsy[0];
        P[35] = asy[0] * asy[0];
#pragma unroll
        for (int i = 0; i < 6; ++i) sX[tid][i] = X[i];
    }
    __syncthreads();

    const _Float16* sAbase = &sA[0][0];
    const int cc = (wv << 4) + (lane & 15);     // this lane's column in [0,128)
    const float bI = biasC[0 * FEATN + cc];
    const float bF = biasC[1 * FEATN + cc];
    const float bG = biasC[2 * FEATN + cc];
    const float bO = biasC[3 * FEATN + cc];

    for (int step = 0; step < TOTAL; ++step) {
        // ---- phase 1: feat = tanh(X @ W_cf^T + b_cf) -> sA f16 ----
        for (int idx = tid; idx < MB * FEATN; idx += NTHREADS) {
            int m = idx / FEATN, j = idx % FEATN;
            float a = b_cf[j];
            const float* wr = &W_cf[j * 6];
#pragma unroll
            for (int k = 0; k < 6; ++k) a += sX[m][k] * wr[k];
            sA[m][j] = (_Float16)tanhf(a);
        }
        __syncthreads();

        // ---- phase 2: WMMA GEMM [32x256] x [256x512] (this wave: 4 gate tiles x 2 M) ----
        v8f acc[2][4];
#pragma unroll
        for (int mt = 0; mt < 2; ++mt)
#pragma unroll
            for (int g = 0; g < 4; ++g) acc[mt][g] = (v8f){0.f,0.f,0.f,0.f,0.f,0.f,0.f,0.f};

#pragma unroll
        for (int kc = 0; kc < KTOT / 32; ++kc) {
            const int k0 = kc * 32;
            v16h a0 = load_a(sAbase, 0, k0, lane);
            v16h a1 = load_a(sAbase, 1, k0, lane);
#pragma unroll
            for (int g = 0; g < 4; ++g) {
                v16h bf = load_b(Wcat, g * FEATN + (wv << 4), k0, lane);
                acc[0][g] = __builtin_amdgcn_wmma_f32_16x16x32_f16(
                    false, a0, false, bf, (short)0, acc[0][g], false, false);
                acc[1][g] = __builtin_amdgcn_wmma_f32_16x16x32_f16(
                    false, a1, false, bf, (short)0, acc[1][g], false, false);
            }
        }
        __syncthreads();   // everyone done reading sA before we write hx back into it

        // ---- phase 3: LSTM pointwise on own accumulators ----
#pragma unroll
        for (int mt = 0; mt < 2; ++mt) {
#pragma unroll
            for (int r = 0; r < 8; ++r) {
                int m = mt * 16 + ((lane >> 4) << 3) + r;
                float iv = acc[mt][0][r] + bI;
                float fv = acc[mt][1][r] + bF;
                float gv = acc[mt][2][r] + bG;
                float ov = acc[mt][3][r] + bO;
                float cn = sigm(fv) * sCx[m][cc] + sigm(iv) * tanhf(gv);
                float hn = sigm(ov) * tanhf(cn);
                sCx[m][cc] = cn;
                sHx[m][cc] = hn;
                sA[m][FEATN + cc] = (_Float16)hn;
            }
        }
        __syncthreads();

        // ---- phase 4: cmd = hx @ W_out^T + b_out  (192 lanes) ----
        if (tid < MB * 6) {
            int m = tid / 6, n = tid % 6;
            float a = b_out[n];
            const float* wr = &W_out[n * FEATN];
            for (int k = 0; k < FEATN; ++k) a += sHx[m][k] * wr[k];
            sCmd[m][n] = a;
        }
        __syncthreads();

        // ---- phase 5: Kalman predict (+update in history / output in pred) ----
        if (tid < MB) {
            const float u0 = sCmd[tid][0], u1 = sCmd[tid][1];
            const float Gm00 = sCmd[tid][2], Gm01 = sCmd[tid][3];
            const float Gm10 = sCmd[tid][4], Gm11 = sCmd[tid][5];

            // Xp = F X + G u   (F sparsity)
            float x0 = X[0] + dt * X[1] + h2 * X[2] + g0c * u0;
            float x1 = X[1] + dt * X[2] + g1c * u0;
            float x2 = X[2] + g2c * u0;
            float x3 = X[3] + dt * X[4] + h2 * X[5] + g0c * u1;
            float x4 = X[4] + dt * X[5] + g1c * u1;
            float x5 = X[5] + g2c * u1;
            X[0] = x0; X[1] = x1; X[2] = x2; X[3] = x3; X[4] = x4; X[5] = x5;

            // Gs = GJ @ Gm
            float Gs[12];
#pragma unroll
            for (int i = 0; i < 6; ++i) {
                Gs[i * 2 + 0] = sGJ[i * 2 + 0] * Gm00 + sGJ[i * 2 + 1] * Gm10;
                Gs[i * 2 + 1] = sGJ[i * 2 + 0] * Gm01 + sGJ[i * 2 + 1] * Gm11;
            }
            // T = F @ P (row combine), P = T @ F^T (col combine) + Q
            float T[36];
#pragma unroll
            for (int j = 0; j < 6; ++j) {
                T[0 * 6 + j] = P[0 * 6 + j] + dt * P[1 * 6 + j] + h2 * P[2 * 6 + j];
                T[1 * 6 + j] = P[1 * 6 + j] + dt * P[2 * 6 + j];
                T[2 * 6 + j] = P[2 * 6 + j];
                T[3 * 6 + j] = P[3 * 6 + j] + dt * P[4 * 6 + j] + h2 * P[5 * 6 + j];
                T[4 * 6 + j] = P[4 * 6 + j] + dt * P[5 * 6 + j];
                T[5 * 6 + j] = P[5 * 6 + j];
            }
#pragma unroll
            for (int i = 0; i < 6; ++i) {
                float c0 = T[i * 6 + 0] + dt * T[i * 6 + 1] + h2 * T[i * 6 + 2];
                float c1 = T[i * 6 + 1] + dt * T[i * 6 + 2];
                float c2 = T[i * 6 + 2];
                float c3 = T[i * 6 + 3] + dt * T[i * 6 + 4] + h2 * T[i * 6 + 5];
                float c4 = T[i * 6 + 4] + dt * T[i * 6 + 5];
                float c5 = T[i * 6 + 5];
                P[i * 6 + 0] = c0; P[i * 6 + 1] = c1; P[i * 6 + 2] = c2;
                P[i * 6 + 3] = c3; P[i * 6 + 4] = c4; P[i * 6 + 5] = c5;
            }
#pragma unroll
            for (int i = 0; i < 6; ++i)
#pragma unroll
                for (int j = 0; j < 6; ++j)
                    P[i * 6 + j] += (Gs[i * 2] * Gs[j * 2] + Gs[i * 2 + 1] * Gs[j * 2 + 1])
                                    * sRho[i * 6 + j];

            if (step < THIST - 1) {
                int b = b0 + tid;
                float zx = hist[((step + 1) * BATCHN + b) * 2 + 0];
                float zy = hist[((step + 1) * BATCHN + b) * 2 + 1];
                float y0 = zx - X[0], y1 = zy - X[3];
                float S00 = P[0]  + sR[0], S01 = P[3]  + sR[1];
                float S10 = P[18] + sR[2], S11 = P[21] + sR[3];
                float inv = 1.0f / (S00 * S11 - S01 * S10);
                float Si00 =  S11 * inv, Si01 = -S01 * inv;
                float Si10 = -S10 * inv, Si11 =  S00 * inv;
                float K0[6], K1[6];
#pragma unroll
                for (int i = 0; i < 6; ++i) {
                    float p0 = P[i * 6 + 0], p3 = P[i * 6 + 3];
                    K0[i] = p0 * Si00 + p3 * Si10;
                    K1[i] = p0 * Si01 + p3 * Si11;
                }
#pragma unroll
                for (int i = 0; i < 6; ++i) X[i] += K0[i] * y0 + K1[i] * y1;
                float T2[36];
#pragma unroll
                for (int i = 0; i < 6; ++i)
#pragma unroll
                    for (int j = 0; j < 6; ++j)
                        T2[i * 6 + j] = P[i * 6 + j] - K0[i] * P[j] - K1[i] * P[18 + j];
#pragma unroll
                for (int i = 0; i < 6; ++i)
#pragma unroll
                    for (int j = 0; j < 6; ++j) {
                        float krk = K0[i] * (sR[0] * K0[j] + sR[1] * K1[j])
                                  + K1[i] * (sR[2] * K0[j] + sR[3] * K1[j]);
                        P[i * 6 + j] = T2[i * 6 + j] - T2[i * 6 + 0] * K0[j]
                                     - T2[i * 6 + 3] * K1[j] + krk;
                    }
            } else {
                int pt = step - (THIST - 1);
                int b  = b0 + tid;
                float sx  = sqrtf(P[0]);
                float sy  = sqrtf(P[21]);
                float rho = (P[3] + P[18]) / (2.0f * sx * sy);
                float* o = out + ((size_t)(pt * BATCHN + b)) * 5;
                o[0] = X[0]; o[1] = X[3]; o[2] = sx; o[3] = sy; o[4] = rho;
            }
#pragma unroll
            for (int i = 0; i < 6; ++i) sX[tid][i] = X[i];
        }
        __syncthreads();
    }
}

extern "C" void kernel_launch(void* const* d_in, const int* in_sizes, int n_in,
                              void* d_out, int out_size, void* d_ws, size_t ws_size,
                              hipStream_t stream) {
    const float* hist  = (const float*)d_in[0];
    const float* psx   = (const float*)d_in[1];
    const float* psy   = (const float*)d_in[2];
    const float* vsx   = (const float*)d_in[3];
    const float* vsy   = (const float*)d_in[4];
    const float* asx   = (const float*)d_in[5];
    const float* asy   = (const float*)d_in[6];
    const float* jerk  = (const float*)d_in[7];
    const float* coefG = (const float*)d_in[8];
    const float* GR    = (const float*)d_in[9];
    const float* W_cf  = (const float*)d_in[10];
    const float* b_cf  = (const float*)d_in[11];
    const float* W_ih  = (const float*)d_in[12];
    const float* W_hh  = (const float*)d_in[13];
    const float* b_ih  = (const float*)d_in[14];
    const float* b_hh  = (const float*)d_in[15];
    const float* W_out = (const float*)d_in[16];
    const float* b_out = (const float*)d_in[17];
    const int*   lenp  = (const int*)d_in[18];
    float* out = (float*)d_out;

    _Float16* Wcat = (_Float16*)d_ws;
    float* biasC   = (float*)((char*)d_ws + (size_t)NGATE * KTOT * sizeof(_Float16));

    kprep_pack<<<(NGATE * KTOT + NTHREADS - 1) / NTHREADS, NTHREADS, 0, stream>>>(
        W_ih, W_hh, b_ih, b_hh, Wcat, biasC);

    kalman_lstm_main<<<BATCHN / MB, NTHREADS, 0, stream>>>(
        hist, psx, psy, vsx, vsy, asx, asy, jerk, coefG, GR,
        W_cf, b_cf, Wcat, biasC, W_out, b_out, lenp, out);
}
